// TopDownNet_16209206575828
// MI455X (gfx1250) — compile-verified, matrix operand order checked by reading, not run on hardware
//
#include <hip/hip_runtime.h>
#include <hip/hip_bf16.h>

typedef __attribute__((ext_vector_type(16))) _Float16 v16h;
typedef __attribute__((ext_vector_type(8)))  _Float16 v8h;
typedef __attribute__((ext_vector_type(8)))  float    v8f;

#define HID      256
#define NIN      14
#define KSTEPS   10
#define CIN_PAD  288          // (256 + 14) padded up to 9*32 for WMMA K-tiles
#define ROWS     64           // rows per block (4 m-tiles of 16)
#define THREADS  256          // 8 wave32s

// Workspace layout in _Float16 elements: transposed f16 weights W^T[256][Kpad]
#define SZ_270   (256 * 288)
#define SZ_256   (256 * 256)
#define OFF_M1T  0
#define OFF_O1T  (SZ_270)
#define OFF_M2T  (2 * SZ_270)
#define OFF_M3T  (OFF_M2T + SZ_256)
#define OFF_O2T  (OFF_M3T + SZ_256)

// ---------------------------------------------------------------------------
// Weight convert + transpose: W[Kin][256] f32  ->  WT[256][Kpad] f16 (zero pad)
// ---------------------------------------------------------------------------
__global__ void weights_to_f16_t(const float* __restrict__ W,
                                 _Float16* __restrict__ WT,
                                 int Kin, int Kpad) {
    int idx = blockIdx.x * blockDim.x + threadIdx.x;
    int total = 256 * Kpad;
    if (idx >= total) return;
    int n = idx / Kpad;
    int k = idx - n * Kpad;
    float v = (k < Kin) ? W[k * 256 + n] : 0.f;
    WT[idx] = (_Float16)v;
}

// ---------------------------------------------------------------------------
// Fused GEMM + bias + ReLU + f16 store-to-LDS.
//   sIn  : LDS activations, [ROWS][LDIN] f16 (row-major)
//   wT   : global f16 transposed weights [256][KT*32]
//   bias : global f32 [256]
//   sOut : LDS output [ROWS][LDOUT] f16 (only first 256 cols written)
// Wave (mtile, nhalf) computes rows [mtile*16, +16), cols [nhalf*128, +128).
//
// Register budget (deliberate): A fragments preloaded once (KT*8 <= 72 VGPRs),
// nt-loop kept ROLLED so only ONE v8f accumulator chain (+ its pipelined B
// fragments) is live at a time -> no spills, stays under 256 VGPRs.
// ---------------------------------------------------------------------------
template <int KT, int LDIN, int LDOUT>
__device__ __forceinline__ void gemm_relu_f16(
    const _Float16* __restrict__ sIn,
    const _Float16* __restrict__ wT,
    const float*    __restrict__ bias,
    _Float16* __restrict__ sOut,
    int mtile, int nhalf, int lane)
{
    const int laneLo = lane & 15;
    const int hiSel  = (lane >> 4) & 1;
    const int rowA   = mtile * 16 + laneLo;
    const int kSel   = hiSel * 8;
    const int rbase  = mtile * 16 + hiSel * 8;

    // Preload all A fragments for this wave's 16 rows (reused by all 8 n-tiles).
    v16h a[KT];
#pragma unroll
    for (int kt = 0; kt < KT; ++kt) {
        const _Float16* ap = sIn + rowA * LDIN + kt * 32 + kSel;
        v8h alo = *(const v8h*)(ap);
        v8h ahi = *(const v8h*)(ap + 16);
        a[kt] = __builtin_shufflevector(alo, ahi,
                0, 1, 2, 3, 4, 5, 6, 7, 8, 9, 10, 11, 12, 13, 14, 15);
    }

#pragma unroll 1   // keep rolled: bounds live registers to one accumulator chain
    for (int nt = 0; nt < 8; ++nt) {
        const int col = nhalf * 128 + nt * 16 + laneLo;
        const _Float16* wcol = wT + col * (KT * 32) + kSel;

        v8f acc = {0.f, 0.f, 0.f, 0.f, 0.f, 0.f, 0.f, 0.f};
#pragma unroll
        for (int kt = 0; kt < KT; ++kt) {
            v8h blo = *(const v8h*)(wcol + kt * 32);
            v8h bhi = *(const v8h*)(wcol + kt * 32 + 16);
            v16h b = __builtin_shufflevector(blo, bhi,
                     0, 1, 2, 3, 4, 5, 6, 7, 8, 9, 10, 11, 12, 13, 14, 15);
            acc = __builtin_amdgcn_wmma_f32_16x16x32_f16(
                /*neg_a=*/false, a[kt], /*neg_b=*/false, b,
                /*c_mod=*/(short)0, acc,
                /*reuse_a=*/false, /*reuse_b=*/false);
        }

        // Epilogue: bias + ReLU + convert to f16, store to LDS.
        // C layout: VGPR j, lanes 0-15 -> M=j ; lanes 16-31 -> M=8+j ; N=lane&15.
        const float bv = bias[col];
#pragma unroll
        for (int j = 0; j < 8; ++j) {
            float v = acc[j] + bv;
            v = v > 0.f ? v : 0.f;
            sOut[(rbase + j) * LDOUT + col] = (_Float16)v;
        }
    }
}

// ---------------------------------------------------------------------------
// Persistent recurrence kernel: each block owns ROWS rows for all K steps.
// ---------------------------------------------------------------------------
__global__ __launch_bounds__(THREADS)
void topdown_kernel(const float* __restrict__ towers,     // [N][K][14]
                    const float* __restrict__ aggregate,  // [256]
                    const _Float16* __restrict__ ws,      // f16 transposed weights
                    const float* __restrict__ M1_b,
                    const float* __restrict__ M2_b,
                    const float* __restrict__ M3_b,
                    const float* __restrict__ O1_b,
                    const float* __restrict__ O2_b,
                    const float* __restrict__ O3_w,       // [256]
                    const float* __restrict__ O3_b,       // [1]
                    float* __restrict__ out,              // [N]
                    int Ntotal)
{
    __shared__ _Float16 sX [ROWS * CIN_PAD];  // summary | tower | zero-pad
    __shared__ _Float16 sH1[ROWS * HID];
    __shared__ _Float16 sH2[ROWS * HID];

    const int tid   = threadIdx.x;
    const int lane  = tid & 31;
    const int wave  = tid >> 5;
    const int mtile = wave & 3;   // 4 M-tiles of 16 rows
    const int nhalf = wave >> 2;  // 2 N-halves of 128 cols
    const int r0    = blockIdx.x * ROWS;

    // Init X: summary = aggregate broadcast, rest zero (pad stays zero forever).
    for (int e = tid; e < ROWS * CIN_PAD; e += THREADS) {
        int c = e % CIN_PAD;
        sX[e] = (c < HID) ? (_Float16)aggregate[c] : (_Float16)0.f;
    }
    float prod = 1.f;
    __syncthreads();

#pragma unroll 1
    for (int s = 0; s < KSTEPS; ++s) {
        const int kidx = KSTEPS - 1 - s;  // flipped tower order

        // Stream this step's tower slice into X[:, 256:270].
        for (int e = tid; e < ROWS * NIN; e += THREADS) {
            int r = e / NIN, c = e - r * NIN;
            int gr = r0 + r;
            float v = (gr < Ntotal)
                        ? towers[((size_t)gr * KSTEPS + kidx) * NIN + c] : 0.f;
            sX[r * CIN_PAD + HID + c] = (_Float16)v;
        }
        __syncthreads();

        // ---- O path (must read X before M3 overwrites it) ----
        gemm_relu_f16<9, CIN_PAD, HID>(sX,  ws + OFF_O1T, O1_b, sH1, mtile, nhalf, lane);
        __syncthreads();
        gemm_relu_f16<8, HID,     HID>(sH1, ws + OFF_O2T, O2_b, sH2, mtile, nhalf, lane);
        __syncthreads();

        // O3: 256 -> 1 dot, sigmoid, running product (thread t owns row t).
        if (tid < ROWS) {
            float ssum = O3_b[0];
            const _Float16* h = sH2 + tid * HID;
            for (int c = 0; c < HID; ++c) ssum += (float)h[c] * O3_w[c];
            prod *= 1.f / (1.f + __expf(-ssum));
        }
        __syncthreads();

        // ---- M path: new summary written back into X[:, 0:256] ----
        gemm_relu_f16<9, CIN_PAD, HID>(sX,  ws + OFF_M1T, M1_b, sH1, mtile, nhalf, lane);
        __syncthreads();
        gemm_relu_f16<8, HID,     HID>(sH1, ws + OFF_M2T, M2_b, sH2, mtile, nhalf, lane);
        __syncthreads();
        gemm_relu_f16<8, HID, CIN_PAD>(sH2, ws + OFF_M3T, M3_b, sX,  mtile, nhalf, lane);
        __syncthreads();
    }

    if (tid < ROWS) {
        int gr = r0 + tid;
        if (gr < Ntotal) out[gr] = prod;
    }
}

// ---------------------------------------------------------------------------
extern "C" void kernel_launch(void* const* d_in, const int* in_sizes, int n_in,
                              void* d_out, int out_size, void* d_ws, size_t ws_size,
                              hipStream_t stream)
{
    const float* towers    = (const float*)d_in[0];
    const float* aggregate = (const float*)d_in[1];
    const float* M1_w = (const float*)d_in[2];
    const float* M1_b = (const float*)d_in[3];
    const float* M2_w = (const float*)d_in[4];
    const float* M2_b = (const float*)d_in[5];
    const float* M3_w = (const float*)d_in[6];
    const float* M3_b = (const float*)d_in[7];
    const float* O1_w = (const float*)d_in[8];
    const float* O1_b = (const float*)d_in[9];
    const float* O2_w = (const float*)d_in[10];
    const float* O2_b = (const float*)d_in[11];
    const float* O3_w = (const float*)d_in[12];
    const float* O3_b = (const float*)d_in[13];

    _Float16* ws = (_Float16*)d_ws;
    const int Ntotal = in_sizes[0] / (KSTEPS * NIN);

    // Convert + transpose weights into workspace (cheap; redone every call
    // so the launch stays deterministic and ws poisoning is harmless).
    const int tpb = 256;
    const int t270 = 256 * 288, t256 = 256 * 256;
    weights_to_f16_t<<<(t270 + tpb - 1) / tpb, tpb, 0, stream>>>(M1_w, ws + OFF_M1T, 270, 288);
    weights_to_f16_t<<<(t270 + tpb - 1) / tpb, tpb, 0, stream>>>(O1_w, ws + OFF_O1T, 270, 288);
    weights_to_f16_t<<<(t256 + tpb - 1) / tpb, tpb, 0, stream>>>(M2_w, ws + OFF_M2T, 256, 256);
    weights_to_f16_t<<<(t256 + tpb - 1) / tpb, tpb, 0, stream>>>(M3_w, ws + OFF_M3T, 256, 256);
    weights_to_f16_t<<<(t256 + tpb - 1) / tpb, tpb, 0, stream>>>(O2_w, ws + OFF_O2T, 256, 256);

    const int nblocks = (Ntotal + ROWS - 1) / ROWS;
    topdown_kernel<<<nblocks, THREADS, 0, stream>>>(
        towers, aggregate, ws,
        M1_b, M2_b, M3_b, O1_b, O2_b, O3_w, O3_b,
        (float*)d_out, Ntotal);
}
